// MultiHeadAttention_63900523430577
// MI455X (gfx1250) — matrix operand order repaired and checked by reading, not compiled, not run on hardware
//
#include <hip/hip_runtime.h>
#include <hip/hip_bf16.h>

// ---------------------------------------------------------------------------
// Problem constants (reference: B=4, L=2048, D=1024, H=16, HD=64)
// ---------------------------------------------------------------------------
constexpr int BB  = 4;
constexpr int LL  = 2048;
constexpr int DD  = 1024;
constexpr int HH  = 16;
constexpr int HDH = 64;            // head dim
constexpr int MM  = BB * LL;       // 8192 rows for the big GEMMs
constexpr float ROPE_BASE = 10000.0f;

typedef __attribute__((ext_vector_type(16))) __bf16        v16bf;
typedef __attribute__((ext_vector_type(8)))  float         v8f;
typedef __attribute__((ext_vector_type(4)))  unsigned int  v4u;
typedef __attribute__((ext_vector_type(4)))  int           v4i;
typedef __attribute__((ext_vector_type(8)))  int           v8i;

// ---------------------------------------------------------------------------
// Helpers
// ---------------------------------------------------------------------------
__device__ __forceinline__ unsigned short f2bf(float f) {
    unsigned int u = __float_as_uint(f);
    unsigned int r = u + 0x7FFFu + ((u >> 16) & 1u);   // round to nearest even
    return (unsigned short)(r >> 16);
}

// Build a v16bf fragment from two contiguous 16-byte runs of bf16 bit patterns.
__device__ __forceinline__ v16bf ld_v16(const unsigned short* p0,
                                        const unsigned short* p1) {
    union { uint4 q[2]; v16bf v; } u;
    u.q[0] = *(const uint4*)p0;
    u.q[1] = *(const uint4*)p1;
    return u.v;
}

__device__ __forceinline__ v8f wmma_bf16(v16bf a, v16bf b, v8f c) {
    // (neg_a, A, neg_b, B, c_mod, C, reuse_a, reuse_b)
    return __builtin_amdgcn_wmma_f32_16x16x32_bf16(false, a, false, b,
                                                   (short)0, c, false, false);
}

__device__ __forceinline__ float redmax16(float v) {
    v = fmaxf(v, __shfl_xor(v, 1, 32));
    v = fmaxf(v, __shfl_xor(v, 2, 32));
    v = fmaxf(v, __shfl_xor(v, 4, 32));
    v = fmaxf(v, __shfl_xor(v, 8, 32));
    return v;                      // reduced over each 16-lane half
}
__device__ __forceinline__ float redsum16(float v) {
    v += __shfl_xor(v, 1, 32);
    v += __shfl_xor(v, 2, 32);
    v += __shfl_xor(v, 4, 32);
    v += __shfl_xor(v, 8, 32);
    return v;
}

// ---------------------------------------------------------------------------
// Tensor Data Mover: DMA a 2D tile (tile_d1 rows x tile_d0 bf16 elements,
// row stride = stride_elems) from global memory into LDS at lds_addr.
// Descriptor layout per CDNA5 ISA. Tracked by TENSORcnt.
// ---------------------------------------------------------------------------
__device__ __forceinline__ void tdm_load_2d_bf16(unsigned lds_addr,
                                                 const void* gptr,
                                                 unsigned tile_d0,
                                                 unsigned tile_d1,
                                                 unsigned tensor_d0,
                                                 unsigned tensor_d1,
                                                 unsigned stride_elems) {
    unsigned long long ga = (unsigned long long)(size_t)gptr;
    v4u g0;
    g0[0] = 1u;                                     // count=1, is_restore=0
    g0[1] = lds_addr;                               // lds_addr [63:32]
    g0[2] = (unsigned)(ga & 0xFFFFFFFFu);           // global_addr[31:0]
    g0[3] = (unsigned)((ga >> 32) & 0x1FFFFFFu)     // global_addr[56:32]
            | (2u << 30);                           // type = 2 ("image")
    v8i g1;
    g1[0] = (int)(1u << 16);                        // data_size = 1 (2 bytes)
    g1[1] = (int)((tensor_d0 & 0xFFFFu) << 16);     // tensor_dim0[15:0] @ bit48
    g1[2] = (int)((tensor_d0 >> 16) | (tensor_d1 << 16));
    g1[3] = (int)((tensor_d1 >> 16) | (tile_d0 << 16));
    g1[4] = (int)(tile_d1 & 0xFFFFu);               // tile_dim1 (tile_dim2=0)
    g1[5] = (int)stride_elems;                      // tensor_dim0_stride[31:0]
    g1[6] = 0;                                      // stride[47:32], dim1_stride lo
    g1[7] = 0;
    v4i z = {0, 0, 0, 0};
#if __clang_major__ >= 23
    v8i z8 = {0, 0, 0, 0, 0, 0, 0, 0};
    __builtin_amdgcn_tensor_load_to_lds(g0, g1, z, z, z8, 0);
#else
    __builtin_amdgcn_tensor_load_to_lds(g0, g1, z, z, 0);
#endif
}

// ---------------------------------------------------------------------------
// Kernel 1: fp32 -> bf16 bit-pattern conversion
// ---------------------------------------------------------------------------
__global__ void k_f32_to_bf16(const float* __restrict__ in,
                              unsigned short* __restrict__ out, int n) {
    int i = blockIdx.x * blockDim.x + threadIdx.x;
    if (i < n) out[i] = f2bf(in[i]);
}

// ---------------------------------------------------------------------------
// Kernel 2: tiled WMMA GEMM  C[M,N] (f32) = A[M,K] (bf16) * Bw[N,K]^T (bf16)
// Block tile 128x128, BK=32, 8 waves; wave tile 32x64 (2x4 WMMA accumulators).
// A/B tiles double-buffered in LDS, filled by the Tensor Data Mover; the next
// tile's DMA overlaps the current tile's 8 WMMAs.
// ---------------------------------------------------------------------------
__global__ __launch_bounds__(256)
void k_gemm_bf16(const unsigned short* __restrict__ A,
                 const unsigned short* __restrict__ Bw,
                 float* __restrict__ C, int M, int N, int K) {
    __shared__ __align__(16) unsigned short As[2][128][32];
    __shared__ __align__(16) unsigned short Bs[2][128][32];

    const int tid  = threadIdx.x;
    const int lane = tid & 31;
    const int w    = tid >> 5;       // 0..7
    const int wm   = w & 3;          // 4 wave-rows of 32
    const int wn   = w >> 2;         // 2 wave-cols of 64
    const int m0   = blockIdx.x * 128;
    const int n0   = blockIdx.y * 128;

    const int kbA  = (lane < 16) ? 0 : 8;   // A-fragment K base (16x32 layout)
    const int kbB  = (lane < 16) ? 0 : 8;   // B-fragment K base (same K order as A)
    const int ln16 = lane & 15;

    const unsigned ldsA0 = (unsigned)(size_t)&As[0][0][0];
    const unsigned ldsA1 = (unsigned)(size_t)&As[1][0][0];
    const unsigned ldsB0 = (unsigned)(size_t)&Bs[0][0][0];
    const unsigned ldsB1 = (unsigned)(size_t)&Bs[1][0][0];

    v8f acc[2][4] = {};

    const int nk = K / 32;
    if (w == 0) {   // prologue: DMA tile 0 into buffer 0 (TDM, EXEC-agnostic)
        tdm_load_2d_bf16(ldsA0, A + (size_t)m0 * K, 32, 128, K, M, K);
        tdm_load_2d_bf16(ldsB0, Bw + (size_t)n0 * K, 32, 128, K, N, K);
    }

    for (int t = 0; t < nk; ++t) {
        const int cur = t & 1;
        if (w == 0) __builtin_amdgcn_s_wait_tensorcnt(0);
        __syncthreads();             // tile `cur` resident; prior reads done

        if (w == 0 && t + 1 < nk) {  // overlap next tile's DMA with compute
            int k1 = (t + 1) * 32;
            tdm_load_2d_bf16(cur ? ldsA0 : ldsA1,
                             A + (size_t)m0 * K + k1, 32, 128, K, M, K);
            tdm_load_2d_bf16(cur ? ldsB0 : ldsB1,
                             Bw + (size_t)n0 * K + k1, 32, 128, K, N, K);
        }

        v16bf af[2], bf[4];
        #pragma unroll
        for (int mt = 0; mt < 2; ++mt) {
            int m = wm * 32 + mt * 16 + ln16;
            af[mt] = ld_v16(&As[cur][m][kbA], &As[cur][m][kbA + 16]);
        }
        #pragma unroll
        for (int nt = 0; nt < 4; ++nt) {
            int n = wn * 64 + nt * 16 + ln16;
            bf[nt] = ld_v16(&Bs[cur][n][kbB], &Bs[cur][n][kbB + 16]);
        }
        #pragma unroll
        for (int mt = 0; mt < 2; ++mt)
            #pragma unroll
            for (int nt = 0; nt < 4; ++nt)
                acc[mt][nt] = wmma_bf16(af[mt], bf[nt], acc[mt][nt]);
    }

    const int rofs = (lane < 16) ? 0 : 8;   // C layout: M = vgpr + 8*(lane>=16)
    #pragma unroll
    for (int mt = 0; mt < 2; ++mt)
        #pragma unroll
        for (int nt = 0; nt < 4; ++nt)
            #pragma unroll
            for (int j = 0; j < 8; ++j) {
                int m = m0 + wm * 32 + mt * 16 + j + rofs;
                int n = n0 + wn * 64 + nt * 16 + ln16;
                C[(size_t)m * N + n] = acc[mt][nt][j];
            }
}

// ---------------------------------------------------------------------------
// Kernel 3: RoPE + reorg.
//  qkv f32 [B, L, 3*D]  ->  Qr,Kr bf16 [B,H,L,HD] (roped), Vt bf16 [B,H,HD,L]
// ---------------------------------------------------------------------------
__global__ void k_rope_reorg(const float* __restrict__ qkv,
                             unsigned short* __restrict__ Qr,
                             unsigned short* __restrict__ Kr,
                             unsigned short* __restrict__ Vt) {
    int idx = blockIdx.x * blockDim.x + threadIdx.x;   // over B*L*H*HD
    int d = idx & (HDH - 1);
    int h = (idx >> 6) & (HH - 1);
    int l = (idx >> 10) & (LL - 1);
    int b = idx >> 21;
    if (b >= BB) return;

    const size_t rowbase = (size_t)(b * LL + l) * (3 * DD);
    // V -> transposed [B,H,HD,L]
    float v = qkv[rowbase + 2 * DD + h * HDH + d];
    Vt[((size_t)(b * HH + h) * HDH + d) * LL + l] = f2bf(v);

    if (d < HDH / 2) {             // rotate pair j = d
        int j = d;
        float inv = __powf(ROPE_BASE, -(2.0f * j) / (float)HDH);
        float ang = (float)l * inv;
        float c = __cosf(ang), s = __sinf(ang);

        size_t qk_out = (size_t)(b * HH + h) * LL * HDH + (size_t)l * HDH;
        float q0 = qkv[rowbase + h * HDH + 2 * j];
        float q1 = qkv[rowbase + h * HDH + 2 * j + 1];
        Qr[qk_out + j]            = f2bf(q0 * c - q1 * s);
        Qr[qk_out + j + HDH / 2]  = f2bf(q0 * s + q1 * c);

        float k0 = qkv[rowbase + DD + h * HDH + 2 * j];
        float k1 = qkv[rowbase + DD + h * HDH + 2 * j + 1];
        Kr[qk_out + j]            = f2bf(k0 * c - k1 * s);
        Kr[qk_out + j + HDH / 2]  = f2bf(k0 * s + k1 * c);
    }
}

// ---------------------------------------------------------------------------
// Kernel 4: flash attention. One wave per (b,h,16-row q tile).
// Per 32-col k chunk: 4 WMMAs for S (K=64), online softmax, 4 WMMAs for PV.
// ---------------------------------------------------------------------------
__global__ __launch_bounds__(32)
void k_flash_attn(const unsigned short* __restrict__ Qr,
                  const unsigned short* __restrict__ Kr,
                  const unsigned short* __restrict__ Vt,
                  unsigned short* __restrict__ attn_out) {
    __shared__ __align__(16) unsigned short Pl[16][32];   // P tile, C->A relayout

    const int lane  = threadIdx.x;
    const int ln16  = lane & 15;
    const int kbA   = (lane < 16) ? 0 : 8;
    const int kbB   = (lane < 16) ? 0 : 8;
    const int rofs  = (lane < 16) ? 0 : 8;

    int id    = blockIdx.x;
    int qtile = id & (LL / 16 - 1);
    int bh    = id >> 11;                 // L/16 = 128 tiles
    int h     = bh & (HH - 1);
    int b     = bh >> 4;

    const unsigned short* Qb = Qr + (size_t)bh * LL * HDH;
    const unsigned short* Kb = Kr + (size_t)bh * LL * HDH;
    const unsigned short* Vb = Vt + (size_t)bh * HDH * LL;

    const float scale = 0.125f;           // HD^-0.5 = 1/8

    // Q A-fragments for Kdim = 64 (two 16x32 fragments), loaded once.
    v16bf qa[2];
    {
        int m = qtile * 16 + ln16;
        #pragma unroll
        for (int s = 0; s < 2; ++s) {
            const unsigned short* p = Qb + (size_t)m * HDH + s * 32 + kbA;
            qa[s] = ld_v16(p, p + 16);
        }
    }

    v8f   o[4] = {};
    float mrow[8], lrow[8];
    #pragma unroll
    for (int j = 0; j < 8; ++j) { mrow[j] = -3.0e38f; lrow[j] = 0.0f; }

    for (int kk0 = 0; kk0 < LL; kk0 += 32) {
        if (kk0 + 32 < LL) {   // hint next K chunk (global_prefetch_b8)
            __builtin_prefetch(Kb + (size_t)(kk0 + 32 + ln16) * HDH, 0, 0);
        }
        // ---- S = Q * K^T for 32 columns ------------------------------------
        v8f s0 = {}, s1 = {};
        #pragma unroll
        for (int s = 0; s < 2; ++s) {
            const unsigned short* p0 =
                Kb + (size_t)(kk0 + ln16) * HDH + s * 32 + kbB;
            const unsigned short* p1 =
                Kb + (size_t)(kk0 + 16 + ln16) * HDH + s * 32 + kbB;
            s0 = wmma_bf16(qa[s], ld_v16(p0, p0 + 16), s0);
            s1 = wmma_bf16(qa[s], ld_v16(p1, p1 + 16), s1);
        }

        // ---- online softmax over the 32 new columns ------------------------
        __syncthreads();
        #pragma unroll
        for (int j = 0; j < 8; ++j) {
            float v0 = s0[j] * scale;
            float v1 = s1[j] * scale;
            float rmax = redmax16(fmaxf(v0, v1));
            float mnew = fmaxf(mrow[j], rmax);
            float alpha = __expf(mrow[j] - mnew);
            float p0 = __expf(v0 - mnew);
            float p1 = __expf(v1 - mnew);
            float rsum = redsum16(p0 + p1);
            lrow[j] = lrow[j] * alpha + rsum;
            mrow[j] = mnew;
            #pragma unroll
            for (int t = 0; t < 4; ++t) o[t][j] *= alpha;
            int r = j + rofs;
            Pl[r][ln16]      = f2bf(p0);
            Pl[r][ln16 + 16] = f2bf(p1);
        }
        __syncthreads();

        // ---- O += P * V  (P as A 16x32, V^T rows contiguous in kk) ---------
        v16bf pa = ld_v16(&Pl[ln16][kbA], &Pl[ln16][kbA + 16]);
        #pragma unroll
        for (int t = 0; t < 4; ++t) {
            const unsigned short* vp =
                Vb + (size_t)(t * 16 + ln16) * LL + kk0 + kbB;
            o[t] = wmma_bf16(pa, ld_v16(vp, vp + 16), o[t]);
        }
    }

    // ---- normalize + scatter to [B, L, D] bf16 -----------------------------
    #pragma unroll
    for (int t = 0; t < 4; ++t)
        #pragma unroll
        for (int j = 0; j < 8; ++j) {
            int q = qtile * 16 + j + rofs;
            int col = h * HDH + t * 16 + ln16;
            float val = o[t][j] / lrow[j];
            attn_out[(size_t)(b * LL + q) * DD + col] = f2bf(val);
        }
}

// ---------------------------------------------------------------------------
// Launch
// ---------------------------------------------------------------------------
extern "C" void kernel_launch(void* const* d_in, const int* in_sizes, int n_in,
                              void* d_out, int out_size, void* d_ws, size_t ws_size,
                              hipStream_t stream) {
    const float* x      = (const float*)d_in[0];   // [B, L, D]
    const float* W_qkv  = (const float*)d_in[1];   // [3D, D]
    const float* W_proj = (const float*)d_in[2];   // [D, D]
    float* out = (float*)d_out;                    // [B, L, D]

    // workspace carve-out (256-B aligned)
    size_t off = 0;
    auto carve = [&](size_t bytes) {
        void* p = (char*)d_ws + off;
        off += (bytes + 255) & ~(size_t)255;
        return p;
    };
    unsigned short* x_bf    = (unsigned short*)carve((size_t)MM * DD * 2);
    unsigned short* wqkv_bf = (unsigned short*)carve((size_t)3 * DD * DD * 2);
    unsigned short* wprj_bf = (unsigned short*)carve((size_t)DD * DD * 2);
    float*          qkv_f   = (float*)carve((size_t)MM * 3 * DD * 4);
    unsigned short* Qr      = (unsigned short*)carve((size_t)MM * DD * 2);
    unsigned short* Kr      = (unsigned short*)carve((size_t)MM * DD * 2);
    unsigned short* Vt      = (unsigned short*)carve((size_t)MM * DD * 2);
    unsigned short* attn_bf = (unsigned short*)carve((size_t)MM * DD * 2);

    // 1) fp32 -> bf16
    {
        int n1 = MM * DD, n2 = 3 * DD * DD, n3 = DD * DD;
        k_f32_to_bf16<<<(n1 + 255) / 256, 256, 0, stream>>>(x, x_bf, n1);
        k_f32_to_bf16<<<(n2 + 255) / 256, 256, 0, stream>>>(W_qkv, wqkv_bf, n2);
        k_f32_to_bf16<<<(n3 + 255) / 256, 256, 0, stream>>>(W_proj, wprj_bf, n3);
    }

    // 2) qkv = x * W_qkv^T   : [8192, 3072]
    k_gemm_bf16<<<dim3(MM / 128, (3 * DD) / 128), 256, 0, stream>>>(
        x_bf, wqkv_bf, qkv_f, MM, 3 * DD, DD);

    // 3) RoPE + reorg
    {
        int n = BB * LL * HH * HDH;
        k_rope_reorg<<<(n + 255) / 256, 256, 0, stream>>>(qkv_f, Qr, Kr, Vt);
    }

    // 4) flash attention
    k_flash_attn<<<BB * HH * (LL / 16), 32, 0, stream>>>(Qr, Kr, Vt, attn_bf);

    // 5) out = attn * W_proj^T : [8192, 1024] f32
    k_gemm_bf16<<<dim3(MM / 128, DD / 128), 256, 0, stream>>>(
        attn_bf, wprj_bf, out, MM, DD, DD);
}
